// Mamba_25108378812674
// MI455X (gfx1250) — matrix operand order repaired and checked by reading, not compiled, not run on hardware
//
#include <hip/hip_runtime.h>

typedef long long i64;
typedef __attribute__((ext_vector_type(16))) __bf16 v16bf;
typedef __attribute__((ext_vector_type(8)))  __bf16 v8bf;
typedef __attribute__((ext_vector_type(8)))  float  v8f;

#define D_MODEL   2048
#define D_INNER   4096
#define D_STATE   128
#define D_HEAD    64
#define N_HEADS   64
#define CONV_DIM  4352      // D_INNER + 2*D_STATE
#define D_IN_PROJ 8512      // 2*D_INNER + 2*D_STATE + N_HEADS
#define BATCH     2
#define SEQ       4096
#define NTOK      (BATCH*SEQ)          // 8192
#define CHUNKLEN  256
#define NCHUNK    16                   // per batch

// ---------------------------------------------------------------------------
// fp32 -> bf16 cast
// ---------------------------------------------------------------------------
__global__ __launch_bounds__(256) void cast_bf16_kernel(const float* __restrict__ in,
                                                        __bf16* __restrict__ out, i64 n) {
  i64 i = (i64)blockIdx.x * 256 + threadIdx.x;
  if (i < n) out[i] = (__bf16)in[i];
}

// ---------------------------------------------------------------------------
// Generic WMMA bf16 GEMM:  C[m,n] = sum_k A[m,k] * B[n,k]
// A: (M x K) row-major bf16, B: (N x K) row-major bf16, C: (M x N) fp32.
// Block = 256 threads = 8 waves laid out 4(M) x 2(N); wave tile = 64x32
// (4 M-tiles x 2 N-tiles -> 8 WMMAs per 12 b128 loads per K-step).
// Requires M%256==0, N%64==0, K%32==0 (true for every call site).
// ---------------------------------------------------------------------------
__global__ __launch_bounds__(256) void gemm_bf16_nt(const __bf16* __restrict__ A,
                                                    const __bf16* __restrict__ B,
                                                    float* __restrict__ C,
                                                    int M, int N, int K,
                                                    i64 sA, i64 sB, i64 sC) {
  (void)M;
  A += (i64)blockIdx.z * sA;
  B += (i64)blockIdx.z * sB;
  C += (i64)blockIdx.z * sC;

  const int lane = threadIdx.x & 31;
  const int wave = threadIdx.x >> 5;
  const int wm = wave & 3, wn = wave >> 2;
  const int m0 = blockIdx.y * 256 + wm * 64;
  const int n0 = blockIdx.x * 64 + wn * 32;
  const int lm = lane & 15;       // row (A) / col (B) within 16-tile
  const int lk = lane >> 4;       // k-half selector

  // Per-lane row base pointers (hoisted out of the K loop)
  const __bf16* arow[4];
  const __bf16* brow[2];
#pragma unroll
  for (int t = 0; t < 4; ++t) arow[t] = A + (i64)(m0 + t * 16 + lm) * K;
#pragma unroll
  for (int t = 0; t < 2; ++t) brow[t] = B + (i64)(n0 + t * 16 + lm) * K;

  v8f acc[4][2] = {};

  for (int k0 = 0; k0 < K; k0 += 32) {
    // Unconditional speculative prefetch ~8 K-steps ahead.
    __builtin_prefetch(arow[0] + k0 + 256, 0, 1);
    __builtin_prefetch(brow[0] + k0 + 256, 0, 1);

    v16bf afr[4], bfr[2];
#pragma unroll
    for (int t = 0; t < 4; ++t) {
      // A fragment: two contiguous runs of 8 bf16 at k0+lk*8 and k0+16+lk*8
      const __bf16* ap = arow[t] + k0 + lk * 8;
      v8bf alo = *(const v8bf*)ap;
      v8bf ahi = *(const v8bf*)(ap + 16);
      afr[t] = __builtin_shufflevector(alo, ahi, 0, 1, 2, 3, 4, 5, 6, 7,
                                       8, 9, 10, 11, 12, 13, 14, 15);
    }
#pragma unroll
    for (int t = 0; t < 2; ++t) {
      // B fragment: two contiguous runs of 8 bf16 at k0+lk*8 and k0+16+lk*8
      const __bf16* bp = brow[t] + k0 + lk * 8;
      v8bf blo = *(const v8bf*)bp;
      v8bf bhi = *(const v8bf*)(bp + 16);
      bfr[t] = __builtin_shufflevector(blo, bhi, 0, 1, 2, 3, 4, 5, 6, 7,
                                       8, 9, 10, 11, 12, 13, 14, 15);
    }
#pragma unroll
    for (int ti = 0; ti < 4; ++ti)
#pragma unroll
      for (int tj = 0; tj < 2; ++tj)
        acc[ti][tj] = __builtin_amdgcn_wmma_f32_16x16x32_bf16(
            false, afr[ti], false, bfr[tj], (short)0, acc[ti][tj], false, false);
  }

#pragma unroll
  for (int ti = 0; ti < 4; ++ti)
#pragma unroll
    for (int tj = 0; tj < 2; ++tj) {
      const int col = n0 + tj * 16 + lm;
#pragma unroll
      for (int v = 0; v < 8; ++v) {
        const int row = m0 + ti * 16 + v + 8 * lk;   // C/D layout: M = v + 8*lanehalf
        C[(i64)row * N + col] = acc[ti][tj][v];
      }
    }
}

// ---------------------------------------------------------------------------
// Depthwise causal conv (width 4) + SiLU over the xBC slice of zxbcdt,
// routing outputs to bf16 x / B / C buffers.
// ---------------------------------------------------------------------------
__global__ __launch_bounds__(256) void conv_silu_kernel(const float* __restrict__ zx,
                                                        const float* __restrict__ conv_w,
                                                        const float* __restrict__ conv_b,
                                                        __bf16* __restrict__ xbf,
                                                        __bf16* __restrict__ Bbf,
                                                        __bf16* __restrict__ Cbf) {
  i64 idx = (i64)blockIdx.x * 256 + threadIdx.x;
  const i64 total = (i64)NTOK * CONV_DIM;
  if (idx >= total) return;
  const int d = (int)(idx % CONV_DIM);
  const i64 tl = idx / CONV_DIM;          // global token
  const int t = (int)(tl % SEQ);
  const i64 bb = tl / SEQ;

  float acc = conv_b[d];
#pragma unroll
  for (int w = 0; w < 4; ++w) {
    const int ts = t + w - 3;
    if (ts >= 0)
      acc += zx[((i64)bb * SEQ + ts) * D_IN_PROJ + D_INNER + d] * conv_w[d * 4 + w];
  }
  acc = acc / (1.f + __expf(-acc));       // SiLU

  if (d < D_INNER)               xbf[tl * D_INNER + d] = (__bf16)acc;
  else if (d < D_INNER + D_STATE) Bbf[tl * D_STATE + (d - D_INNER)] = (__bf16)acc;
  else                            Cbf[tl * D_STATE + (d - D_INNER - D_STATE)] = (__bf16)acc;
}

// ---------------------------------------------------------------------------
// dt = softplus(dt_raw + dt_bias); dA = -exp(A_log)*dt; inclusive cumsum per
// chunk. One block per (batch, head, chunk). Layout: [(bb*64+h)*16+c][256].
// ---------------------------------------------------------------------------
__global__ __launch_bounds__(256) void dt_scan_kernel(const float* __restrict__ zx,
                                                      const float* __restrict__ dt_bias,
                                                      const float* __restrict__ A_log,
                                                      float* __restrict__ dts,
                                                      float* __restrict__ dAcs) {
  const int bid = blockIdx.x;             // ((bb*64 + h)*16 + c)
  const int c = bid & 15;
  const int h = (bid >> 4) & 63;
  const int bb = bid >> 10;
  const int t = threadIdx.x;

  const i64 tok = (i64)bb * SEQ + c * CHUNKLEN + t;
  float raw = zx[tok * D_IN_PROJ + (D_INNER + CONV_DIM) + h] + dt_bias[h];
  float sp = raw > 20.f ? raw : log1pf(__expf(raw));
  float dA = -__expf(A_log[h]) * sp;

  __shared__ float sc[CHUNKLEN];
  sc[t] = dA;
  __syncthreads();
  for (int o = 1; o < CHUNKLEN; o <<= 1) {
    float v = (t >= o) ? sc[t - o] : 0.f;
    __syncthreads();
    sc[t] += v;
    __syncthreads();
  }
  const i64 o = (i64)bid * CHUNKLEN + t;
  dts[o] = sp;
  dAcs[o] = sc[t];
}

// ---------------------------------------------------------------------------
// Chunked SSM: one block per (batch, head). Sequential over 16 chunks with
// 64x128 fp32 state in LDS. y_off and y_diag use bf16 WMMA; state update is
// a small VALU GEMM. 8 waves; wave w owns output rows [32w, 32w+32).
// ---------------------------------------------------------------------------
__global__ __launch_bounds__(256) void ssm_kernel(const __bf16* __restrict__ xbf,   // (NTOK, D_INNER)
                                                  const __bf16* __restrict__ Bbf,   // (NTOK, D_STATE)
                                                  const __bf16* __restrict__ Cbf,   // (NTOK, D_STATE)
                                                  const float* __restrict__ CBbuf,  // (B*16, 256, 256)
                                                  const float* __restrict__ dAcs,   // ((bb*64+h)*16+c, 256)
                                                  const float* __restrict__ dts,
                                                  const float* __restrict__ Dp,
                                                  float* __restrict__ y) {          // (NTOK, D_INNER)
  extern __shared__ __align__(32) char smem[];
  __bf16* xT  = (__bf16*)smem;                      // [64][256]  x transposed (p, j)
  __bf16* Bl  = (__bf16*)(smem + 32768);            // [256][128] B chunk (j, n)
  __bf16* sbf = (__bf16*)(smem + 98304);            // [64][128]  state bf16 (p, n)
  float*  sst = (float*)(smem + 114688);            // [64][128]  state fp32
  float*  sA  = (float*)(smem + 147456);            // [256] dA cumsum
  float*  sdt = (float*)(smem + 148480);            // [256] dt
  float*  sw  = (float*)(smem + 149504);            // [256] exp(last - dAcs)*dt

  const int bh = blockIdx.x;
  const int bb = bh >> 6, h = bh & 63;
  const int tid = threadIdx.x;
  const int lane = tid & 31, wave = tid >> 5;
  const int lm = lane & 15, lk = lane >> 4;
  const float dcoef = Dp[h];

  for (int e = tid; e < D_HEAD * D_STATE; e += 256) sst[e] = 0.f;
  __syncthreads();

  for (int c = 0; c < NCHUNK; ++c) {
    const i64 tokbase = (i64)bb * SEQ + c * CHUNKLEN;

    // ---- stage chunk data into LDS ----
    {
      const i64 dbase = ((i64)(bb * 64 + h) * 16 + c) * CHUNKLEN;
      sA[tid] = dAcs[dbase + tid];
      sdt[tid] = dts[dbase + tid];
      __syncthreads();
      const float last = sA[CHUNKLEN - 1];
      sw[tid] = __expf(last - sA[tid]) * sdt[tid];
      for (int e = tid; e < D_HEAD * D_STATE; e += 256) sbf[e] = (__bf16)sst[e];
      for (int e = tid; e < D_HEAD * CHUNKLEN; e += 256) {
        const int p = e >> 8, j = e & 255;
        xT[p * CHUNKLEN + j] = xbf[(tokbase + j) * D_INNER + h * D_HEAD + p];
      }
      for (int e = tid; e < CHUNKLEN * D_STATE; e += 256) {
        const int j = e >> 7, n = e & 127;
        Bl[j * D_STATE + n] = Bbf[(tokbase + j) * D_STATE + n];
      }
      __syncthreads();
    }

    // ---- y = y_off + y_diag via WMMA ----
    v8f acc[2][4] = {};

    // y_off: (exp(dAcs_i)*C[i,:]) @ state^T; K over D_STATE
    for (int n0 = 0; n0 < D_STATE; n0 += 32) {
      v16bf afr[2], bfr[4];
#pragma unroll
      for (int t = 0; t < 2; ++t) {
        const int i = wave * 32 + t * 16 + lm;
        const float ei = __expf(sA[i]);
        const __bf16* cp = Cbf + (tokbase + i) * D_STATE + n0 + lk * 8;
        v8bf c0 = *(const v8bf*)cp;
        v8bf c1 = *(const v8bf*)(cp + 16);
        v16bf a;
#pragma unroll
        for (int q = 0; q < 8; ++q) {
          a[q] = (__bf16)((float)c0[q] * ei);
          a[q + 8] = (__bf16)((float)c1[q] * ei);
        }
        afr[t] = a;
      }
#pragma unroll
      for (int pj = 0; pj < 4; ++pj)
        bfr[pj] = __builtin_shufflevector(*(const v8bf*)&sbf[(pj * 16 + lm) * D_STATE + n0 + lk * 8], *(const v8bf*)&sbf[(pj * 16 + lm) * D_STATE + n0 + 16 + lk * 8], 0, 1, 2, 3, 4, 5, 6, 7, 8, 9, 10, 11, 12, 13, 14, 15);
#pragma unroll
      for (int t = 0; t < 2; ++t)
#pragma unroll
        for (int pj = 0; pj < 4; ++pj)
          acc[t][pj] = __builtin_amdgcn_wmma_f32_16x16x32_bf16(
              false, afr[t], false, bfr[pj], (short)0, acc[t][pj], false, false);
    }

    // y_diag: M @ x, M[i,j] = CB[i,j]*exp(dAcs_i-dAcs_j)*dt_j, j<=i
    for (int j0 = 0; j0 <= wave * 32; j0 += 32) {
      v16bf afr[2], bfr[4];
#pragma unroll
      for (int t = 0; t < 2; ++t) {
        const int i = wave * 32 + t * 16 + lm;
        const float dAi = sA[i];
        const float* cb = CBbuf + ((i64)(bb * 16 + c) * CHUNKLEN + i) * CHUNKLEN + j0 + lk * 8;
        v8f cb0 = *(const v8f*)cb;
        v8f cb1 = *(const v8f*)(cb + 16);
        v16bf a;
#pragma unroll
        for (int q = 0; q < 8; ++q) {
          const int j1 = j0 + lk * 8 + q;
          const int j2 = j0 + 16 + lk * 8 + q;
          float m1 = (j1 <= i) ? cb0[q] * __expf(dAi - sA[j1]) * sdt[j1] : 0.f;
          float m2 = (j2 <= i) ? cb1[q] * __expf(dAi - sA[j2]) * sdt[j2] : 0.f;
          a[q] = (__bf16)m1;
          a[q + 8] = (__bf16)m2;
        }
        afr[t] = a;
      }
#pragma unroll
      for (int pj = 0; pj < 4; ++pj)
        bfr[pj] = __builtin_shufflevector(*(const v8bf*)&xT[(pj * 16 + lm) * CHUNKLEN + j0 + lk * 8], *(const v8bf*)&xT[(pj * 16 + lm) * CHUNKLEN + j0 + 16 + lk * 8], 0, 1, 2, 3, 4, 5, 6, 7, 8, 9, 10, 11, 12, 13, 14, 15);
#pragma unroll
      for (int t = 0; t < 2; ++t)
#pragma unroll
        for (int pj = 0; pj < 4; ++pj)
          acc[t][pj] = __builtin_amdgcn_wmma_f32_16x16x32_bf16(
              false, afr[t], false, bfr[pj], (short)0, acc[t][pj], false, false);
    }

    // ---- write y = acc + D*x ----
#pragma unroll
    for (int t = 0; t < 2; ++t)
#pragma unroll
      for (int pj = 0; pj < 4; ++pj)
#pragma unroll
        for (int v = 0; v < 8; ++v) {
          const int ic = wave * 32 + t * 16 + v + 8 * lk;
          const int p = pj * 16 + lm;
          const float xv = (float)xT[p * CHUNKLEN + ic];
          y[(tokbase + ic) * D_INNER + h * D_HEAD + p] = acc[t][pj][v] + dcoef * xv;
        }

    // ---- state update: s = s*exp(dAcs_last) + (w*x)^T @ B ----
    const float cdec = __expf(sA[CHUNKLEN - 1]);
    for (int e = tid; e < D_HEAD * D_STATE; e += 256) {
      const int p = e >> 7, n = e & 127;
      float a = 0.f;
      for (int j = 0; j < CHUNKLEN; ++j)
        a += sw[j] * (float)xT[p * CHUNKLEN + j] * (float)Bl[j * D_STATE + n];
      sst[e] = sst[e] * cdec + a;
    }
    __syncthreads();
  }
}

// ---------------------------------------------------------------------------
// Gating (y * silu(z)) + RMSNorm + norm_w, output bf16 for out_proj GEMM.
// One block per token.
// ---------------------------------------------------------------------------
__global__ __launch_bounds__(256) void gate_norm_kernel(const float* __restrict__ zx,
                                                        const float* __restrict__ y,
                                                        const float* __restrict__ norm_w,
                                                        __bf16* __restrict__ ybf) {
  const i64 tok = blockIdx.x;
  const float* zrow = zx + tok * D_IN_PROJ;
  const float* yrow = y + tok * D_INNER;
  const int tid = threadIdx.x, lane = tid & 31, wave = tid >> 5;

  float g[16];
  float ss = 0.f;
#pragma unroll
  for (int t = 0; t < 16; ++t) {
    const int d = t * 256 + tid;
    const float z = zrow[d];
    const float gv = yrow[d] * z / (1.f + __expf(-z));
    g[t] = gv;
    ss += gv * gv;
  }
#pragma unroll
  for (int o = 16; o > 0; o >>= 1) ss += __shfl_xor(ss, o);

  __shared__ float red[8];
  if (lane == 0) red[wave] = ss;
  __syncthreads();
  if (wave == 0) {
    float tot = (lane < 8) ? red[lane] : 0.f;
#pragma unroll
    for (int o = 4; o > 0; o >>= 1) tot += __shfl_xor(tot, o);
    if (lane == 0) red[0] = tot;
  }
  __syncthreads();
  const float r = rsqrtf(red[0] * (1.f / D_INNER) + 1e-5f);
#pragma unroll
  for (int t = 0; t < 16; ++t) {
    const int d = t * 256 + tid;
    ybf[tok * D_INNER + d] = (__bf16)(g[t] * r * norm_w[d]);
  }
}

// ---------------------------------------------------------------------------
// Host-side orchestration
// ---------------------------------------------------------------------------
extern "C" void kernel_launch(void* const* d_in, const int* in_sizes, int n_in,
                              void* d_out, int out_size, void* d_ws, size_t ws_size,
                              hipStream_t stream) {
  (void)in_sizes; (void)n_in; (void)out_size; (void)ws_size;

  const float* u       = (const float*)d_in[0];
  const float* W_in    = (const float*)d_in[1];
  const float* conv_w  = (const float*)d_in[2];
  const float* conv_b  = (const float*)d_in[3];
  const float* dt_bias = (const float*)d_in[4];
  const float* A_log   = (const float*)d_in[5];
  const float* Dp      = (const float*)d_in[6];
  const float* norm_w  = (const float*)d_in[7];
  const float* W_out   = (const float*)d_in[8];
  float* out = (float*)d_out;

  char* w = (char*)d_ws;
  auto carve = [&](size_t bytes) -> void* {
    void* p = (void*)w;
    w += (bytes + 255) & ~(size_t)255;
    return p;
  };

  __bf16* u_bf    = (__bf16*)carve((size_t)NTOK * D_MODEL * 2);
  __bf16* Win_bf  = (__bf16*)carve((size_t)D_IN_PROJ * D_MODEL * 2);
  __bf16* Wout_bf = (__bf16*)carve((size_t)D_MODEL * D_INNER * 2);
  float*  zx      = (float*)carve((size_t)NTOK * D_IN_PROJ * 4);
  __bf16* x_bf    = (__bf16*)carve((size_t)NTOK * D_INNER * 2);
  __bf16* B_bf    = (__bf16*)carve((size_t)NTOK * D_STATE * 2);
  __bf16* C_bf    = (__bf16*)carve((size_t)NTOK * D_STATE * 2);
  float*  dts     = (float*)carve((size_t)BATCH * N_HEADS * NCHUNK * CHUNKLEN * 4);
  float*  dAcs    = (float*)carve((size_t)BATCH * N_HEADS * NCHUNK * CHUNKLEN * 4);
  float*  CBbuf   = (float*)carve((size_t)BATCH * NCHUNK * CHUNKLEN * CHUNKLEN * 4);
  float*  ybuf    = (float*)carve((size_t)NTOK * D_INNER * 4);
  __bf16* yg_bf   = (__bf16*)carve((size_t)NTOK * D_INNER * 2);

  // 1) casts to bf16
  {
    i64 n;
    n = (i64)NTOK * D_MODEL;
    cast_bf16_kernel<<<(unsigned)((n + 255) / 256), 256, 0, stream>>>(u, u_bf, n);
    n = (i64)D_IN_PROJ * D_MODEL;
    cast_bf16_kernel<<<(unsigned)((n + 255) / 256), 256, 0, stream>>>(W_in, Win_bf, n);
    n = (i64)D_MODEL * D_INNER;
    cast_bf16_kernel<<<(unsigned)((n + 255) / 256), 256, 0, stream>>>(W_out, Wout_bf, n);
  }

  // 2) in_proj: zx (8192 x 8512) = u_bf (8192 x 2048) @ W_in^T
  gemm_bf16_nt<<<dim3(D_IN_PROJ / 64, NTOK / 256, 1), 256, 0, stream>>>(
      u_bf, Win_bf, zx, NTOK, D_IN_PROJ, D_MODEL, 0, 0, 0);

  // 3) conv + silu + split
  {
    const i64 n = (i64)NTOK * CONV_DIM;
    conv_silu_kernel<<<(unsigned)((n + 255) / 256), 256, 0, stream>>>(
        zx, conv_w, conv_b, x_bf, B_bf, C_bf);
  }

  // 4) dt softplus + per-chunk cumsum
  dt_scan_kernel<<<BATCH * N_HEADS * NCHUNK, 256, 0, stream>>>(
      zx, dt_bias, A_log, dts, dAcs);

  // 5) CB[i,j] = C[i,:] . B[j,:] per (batch, chunk) — batched WMMA GEMM
  gemm_bf16_nt<<<dim3(CHUNKLEN / 64, CHUNKLEN / 256 + (CHUNKLEN % 256 ? 1 : 0), BATCH * NCHUNK),
                256, 0, stream>>>(
      C_bf, B_bf, CBbuf, CHUNKLEN, CHUNKLEN, D_STATE,
      (i64)CHUNKLEN * D_STATE, (i64)CHUNKLEN * D_STATE, (i64)CHUNKLEN * CHUNKLEN);

  // 6) chunked SSM scan
  {
    const int smem_bytes = 150528;
    (void)hipFuncSetAttribute((const void*)ssm_kernel,
                              hipFuncAttributeMaxDynamicSharedMemorySize, smem_bytes);
    ssm_kernel<<<BATCH * N_HEADS, 256, smem_bytes, stream>>>(
        x_bf, B_bf, C_bf, CBbuf, dAcs, dts, Dp, ybuf);
  }

  // 7) gating + RMSNorm
  gate_norm_kernel<<<NTOK, 256, 0, stream>>>(zx, ybuf, norm_w, yg_bf);

  // 8) out_proj: out (8192 x 2048) = yg_bf (8192 x 4096) @ W_out^T
  gemm_bf16_nt<<<dim3(D_MODEL / 64, NTOK / 256, 1), 256, 0, stream>>>(
      yg_bf, Wout_bf, out, NTOK, D_MODEL, D_INNER, 0, 0, 0);
}
